// GSSM_v4_35605278884493
// MI455X (gfx1250) — compile-verified, matrix operand order checked by reading, not run on hardware
//
#include <hip/hip_runtime.h>
#include <hip/hip_bf16.h>
#include <math.h>

// ---------------- problem constants ----------------
constexpr int Bq   = 4;
constexpr int Lq   = 16384;
constexpr int Dq   = 128;
constexpr int DIq  = 256;
constexpr int DSq  = 16;
constexpr int DTRq = 8;
constexpr int Kc   = 4;           // conv taps
constexpr int Nq   = Bq * Lq;     // 65536 tokens
constexpr int CH   = 256;         // scan chunk length
constexpr int NC   = Lq / CH;     // 64 chunks per batch
constexpr float EPSq = 1e-5f;

typedef __attribute__((ext_vector_type(16))) _Float16 v16h;
typedef __attribute__((ext_vector_type(8)))  _Float16 v8h;
typedef __attribute__((ext_vector_type(8)))  float    v8f;

// =====================================================================
// K0a: pack an fp32 row-major (Kdim x Ncols) weight into the WMMA
// B-operand lane layout for v_wmma_f32_16x16x32_f16.
// packed[((nt*nKB + kb)*32 + lane)*16 + i]
//   lane<16 : column n = nt*16+lane, K-local = {0..7,16..23}
//   lane>=16: same column set, K-local = {8..15,24..31}
// =====================================================================
__global__ void k_pack_b(const float* __restrict__ w, _Float16* __restrict__ packed,
                         int Ncols, int nKB, int total) {
    int tid = blockIdx.x * 256 + threadIdx.x;
    if (tid >= total) return;
    int i    = tid & 15;
    int lane = (tid >> 4) & 31;
    int grp  = tid >> 9;            // nt*nKB + kb
    int kb   = grp % nKB;
    int nt   = grp / nKB;
    int n    = nt * 16 + (lane & 15);
    int sel  = lane >> 4;
    int kl   = (i < 8) ? (i + sel * 8) : (i - 8 + 16 + sel * 8);
    int k    = kb * 32 + kl;
    packed[tid] = (_Float16)w[(size_t)k * Ncols + n];
}

// K0b: Aneg[d*16+s] = -exp(A_log[d,s])
__global__ void k_aneg(const float* __restrict__ a_log, float* __restrict__ aneg) {
    int t = blockIdx.x * 256 + threadIdx.x;
    if (t < DIq * DSq) aneg[t] = -__expf(a_log[t]);
}

// =====================================================================
// K1: pe = coords@pos_w + pos_b ; gather by perm ; RMSNorm*rms_w -> f16
// one block (128 threads) per output token
// =====================================================================
__global__ void k_prep(const float* __restrict__ feats,
                       const float* __restrict__ pos_w, const float* __restrict__ pos_b,
                       const float* __restrict__ rms_w,
                       const int* __restrict__ coords, const int* __restrict__ perm,
                       _Float16* __restrict__ xf16) {
    int tok = blockIdx.x;            // b*L + l (permuted order)
    int d   = threadIdx.x;           // 0..127
    int b   = tok >> 14;
    int src = perm[tok];             // perm[b,l]
    size_t srow = (size_t)(b << 14) + src;
    float cx = (float)coords[srow * 3 + 0];
    float cy = (float)coords[srow * 3 + 1];
    float cz = (float)coords[srow * 3 + 2];
    float pe = cx * pos_w[d] + cy * pos_w[Dq + d] + cz * pos_w[2 * Dq + d] + pos_b[d];
    float v  = feats[srow * Dq + d] + pe;

    // block reduction of v*v over 128 threads (4 waves)
    float ss = v * v;
    #pragma unroll
    for (int off = 16; off > 0; off >>= 1) ss += __shfl_xor(ss, off, 32);
    __shared__ float red[4];
    if ((threadIdx.x & 31) == 0) red[threadIdx.x >> 5] = ss;
    __syncthreads();
    float tot = red[0] + red[1] + red[2] + red[3];
    float rs  = rsqrtf(tot * (1.0f / Dq) + EPSq);
    xf16[(size_t)tok * Dq + d] = (_Float16)(v * rs * rms_w[d]);
}

// =====================================================================
// K2: xz = x @ in_proj_w    (M=65536, K=128, N=512)  via WMMA f16->f32
// block = 256 thr = 8 waves; A tile staged once in LDS, shared by waves.
// wave w -> ntile = blockIdx.y*8+w (uniform xi/z split per wave)
// =====================================================================
__global__ void k_gemm_in(const _Float16* __restrict__ xA,
                          const _Float16* __restrict__ packB,
                          float* __restrict__ xi, float* __restrict__ zbuf) {
    __shared__ _Float16 sA[16 * 128];               // 4 KB A tile
    int tid   = threadIdx.x;
    int lane  = tid & 31;
    int wave  = tid >> 5;
    int mtile = blockIdx.x;
    int ntile = blockIdx.y * 8 + wave;
    int sel   = lane >> 4;

    // cooperative A-tile stage: 256 thr x 16B
    {
        int r = tid >> 4;                           // row 0..15
        int c = (tid & 15) * 8;                     // col halves
        *(v8h*)&sA[r * 128 + c] =
            *(const v8h*)&xA[((size_t)mtile * 16 + r) * 128 + c];
    }
    __syncthreads();

    const _Float16* brow = packB + ((size_t)ntile * 4 * 32 + lane) * 16;
    v8f acc = {};
    #pragma unroll
    for (int kb = 0; kb < 4; ++kb) {
        const _Float16* ap = &sA[(lane & 15) * 128 + kb * 32 + sel * 8];
        if (kb < 3) __builtin_prefetch(brow + (size_t)(kb + 1) * 32 * 16, 0, 3);
        v8h alo = *(const v8h*)(ap);
        v8h ahi = *(const v8h*)(ap + 16);
        v16h a  = __builtin_shufflevector(alo, ahi, 0,1,2,3,4,5,6,7,8,9,10,11,12,13,14,15);
        v16h bm = *(const v16h*)(brow + (size_t)kb * 32 * 16);
        acc = __builtin_amdgcn_wmma_f32_16x16x32_f16(false, a, false, bm,
                                                     (short)0, acc, false, false);
    }

    // wave-uniform destination select (no per-lane exec juggling)
    float* dst;
    int col0;
    if (ntile < 16) { dst = xi;   col0 = ntile * 16; }
    else            { dst = zbuf; col0 = (ntile - 16) * 16; }
    int col = col0 + (lane & 15);
    #pragma unroll
    for (int r = 0; r < 8; ++r) {
        int rrow = mtile * 16 + sel * 8 + r;
        dst[(size_t)rrow * DIq + col] = acc[r];
    }
}

// =====================================================================
// K3: causal depthwise conv (K=4) + bias + SiLU -> xc
// =====================================================================
__global__ void k_conv(const float* __restrict__ xi, const float* __restrict__ cw,
                       const float* __restrict__ cb, float* __restrict__ xc) {
    int tok = blockIdx.x;
    int d   = threadIdx.x;
    int b   = tok >> 14;
    int l   = tok & (Lq - 1);
    float acc = cb[d];
    #pragma unroll
    for (int j = 0; j < Kc; ++j) {
        int ls = l - (Kc - 1) + j;
        if (ls >= 0)
            acc += xi[((size_t)(b << 14) + ls) * DIq + d] * cw[d * Kc + j];
    }
    acc = acc / (1.0f + __expf(-acc));        // SiLU
    xc[(size_t)tok * DIq + d] = acc;
}

// =====================================================================
// K4: dbc = xc @ x_proj_w (256->40); dt = softplus(dbc[:8]@dt_proj_w + b)
//      Bm = dbc[8:24], Cm = dbc[24:40].  One block / token, 256 threads.
// dbc parallelized over 4-lane groups (64 MACs + 2 shfl per output).
// =====================================================================
__global__ void k_xproj(const float* __restrict__ xc, const float* __restrict__ xw,
                        const float* __restrict__ dtw, const float* __restrict__ dtb,
                        float* __restrict__ dt, float* __restrict__ Bm,
                        float* __restrict__ Cm) {
    __shared__ float sx[DIq];
    __shared__ float sdbc[40];
    int tok = blockIdx.x;
    int tid = threadIdx.x;
    sx[tid] = xc[(size_t)tok * DIq + tid];
    __syncthreads();
    if (tid < 160) {
        int j = tid >> 2;                 // output 0..39
        int g = tid & 3;                  // 4-lane group slice
        float a = 0.f;
        #pragma unroll 8
        for (int k = g * 64; k < g * 64 + 64; ++k) a += sx[k] * xw[k * 40 + j];
        a += __shfl_xor(a, 1, 32);
        a += __shfl_xor(a, 2, 32);
        if (g == 0) sdbc[j] = a;
    }
    __syncthreads();
    float acc = dtb[tid];
    #pragma unroll
    for (int r = 0; r < DTRq; ++r) acc += sdbc[r] * dtw[r * DIq + tid];
    float dtv = (acc > 20.f) ? acc : log1pf(__expf(acc));   // softplus
    dt[(size_t)tok * DIq + tid] = dtv;
    if (tid < DSq)                Bm[(size_t)tok * DSq + tid]        = sdbc[8 + tid];
    else if (tid < 2 * DSq)       Cm[(size_t)tok * DSq + (tid - 16)] = sdbc[24 + (tid - 16)];
}

// =====================================================================
// K5: scan phase 1 — per-chunk local scan. block=(b,chunk), thread=d.
// =====================================================================
__global__ void k_scan1(const float* __restrict__ dt, const float* __restrict__ xc,
                        const float* __restrict__ Bm, const float* __restrict__ aneg,
                        float* __restrict__ csa, float* __restrict__ csh) {
    __shared__ float sB[CH * DSq];
    int bc = blockIdx.x;             // b*NC + c
    int d  = threadIdx.x;
    size_t tbase = (size_t)(bc) * CH;           // token base = b*L + c*CH
    for (int i = d; i < CH * DSq; i += 256) sB[i] = Bm[tbase * DSq + i];
    __syncthreads();

    float aA[DSq], ap[DSq], hl[DSq];
    #pragma unroll
    for (int s = 0; s < DSq; ++s) { aA[s] = aneg[d * DSq + s]; ap[s] = 1.f; hl[s] = 0.f; }

    for (int t = 0; t < CH; ++t) {
        size_t n = tbase + t;
        float dtv = dt[n * DIq + d];
        float xv  = xc[n * DIq + d];
        float bf  = dtv * xv;
        #pragma unroll
        for (int s = 0; s < DSq; ++s) {
            float a = __expf(dtv * aA[s]);
            ap[s] *= a;
            hl[s]  = a * hl[s] + bf * sB[t * DSq + s];
        }
    }
    size_t o = ((size_t)bc * DIq + d) * DSq;
    #pragma unroll
    for (int s = 0; s < DSq; ++s) { csa[o + s] = ap[s]; csh[o + s] = hl[s]; }
}

// =====================================================================
// K6: scan phase 2 — sequential combine over 64 chunks; thread=(b,d,s).
// =====================================================================
__global__ void k_scan2(const float* __restrict__ csa, const float* __restrict__ csh,
                        float* __restrict__ cstart) {
    int flat = blockIdx.x * 256 + threadIdx.x;          // b*(DI*DS) + r
    int b = flat / (DIq * DSq);
    int r = flat - b * (DIq * DSq);
    float st = 0.f;
    for (int c = 0; c < NC; ++c) {
        size_t o = ((size_t)(b * NC + c) * (DIq * DSq)) + r;
        cstart[o] = st;
        st = csa[o] * st + csh[o];
    }
}

// =====================================================================
// K7: scan phase 3 — replay with carried-in state; fuse +xc*D, *silu(z),
// emit f16 for the output GEMM.
// =====================================================================
__global__ void k_scan3(const float* __restrict__ dt, const float* __restrict__ xc,
                        const float* __restrict__ Bm, const float* __restrict__ Cm,
                        const float* __restrict__ aneg, const float* __restrict__ cstart,
                        const float* __restrict__ Dp, const float* __restrict__ zbuf,
                        _Float16* __restrict__ y16) {
    __shared__ float sB[CH * DSq];
    __shared__ float sC[CH * DSq];
    int bc = blockIdx.x;
    int d  = threadIdx.x;
    size_t tbase = (size_t)bc * CH;
    for (int i = d; i < CH * DSq; i += 256) {
        sB[i] = Bm[tbase * DSq + i];
        sC[i] = Cm[tbase * DSq + i];
    }
    __syncthreads();

    float aA[DSq], h[DSq];
    size_t so = ((size_t)bc * DIq + d) * DSq;
    #pragma unroll
    for (int s = 0; s < DSq; ++s) { aA[s] = aneg[d * DSq + s]; h[s] = cstart[so + s]; }
    float Dpv = Dp[d];

    for (int t = 0; t < CH; ++t) {
        size_t n = tbase + t;
        float dtv = dt[n * DIq + d];
        float xv  = xc[n * DIq + d];
        float bf  = dtv * xv;
        float y   = 0.f;
        #pragma unroll
        for (int s = 0; s < DSq; ++s) {
            float a = __expf(dtv * aA[s]);
            h[s] = a * h[s] + bf * sB[t * DSq + s];
            y   += h[s] * sC[t * DSq + s];
        }
        float yv = y + xv * Dpv;
        float zv = zbuf[n * DIq + d];
        yv *= zv / (1.0f + __expf(-zv));          // * silu(z)
        y16[n * DIq + d] = (_Float16)yv;
    }
}

// =====================================================================
// K8: out = y @ out_proj_w (M=65536, K=256, N=128) via WMMA; A tile
// staged in LDS; fused LayerNorm over the 128-wide row + perm scatter.
// block = 256 thr = 8 waves covering the full N=128 for one 16-row tile.
// =====================================================================
__global__ void k_gemm_out(const _Float16* __restrict__ y16,
                           const _Float16* __restrict__ packB,
                           const int* __restrict__ perm,
                           const float* __restrict__ lnw, const float* __restrict__ lnb,
                           float* __restrict__ dout) {
    __shared__ _Float16 sA[16 * 256];               // 8 KB A tile
    __shared__ float    sout[16 * 128];             // 8 KB result tile
    int tid   = threadIdx.x;
    int lane  = tid & 31;
    int wave  = tid >> 5;    // = ntile 0..7
    int mtile = blockIdx.x;
    int sel   = lane >> 4;

    // cooperative A-tile stage: 256 thr x 32B
    {
        int r = tid >> 4;                           // row 0..15
        int c = (tid & 15) * 16;                    // col halves
        const _Float16* src = &y16[((size_t)mtile * 16 + r) * 256 + c];
        *(v8h*)&sA[r * 256 + c]     = *(const v8h*)(src);
        *(v8h*)&sA[r * 256 + c + 8] = *(const v8h*)(src + 8);
    }
    __syncthreads();

    const _Float16* brow = packB + ((size_t)wave * 8 * 32 + lane) * 16;
    v8f acc = {};
    #pragma unroll
    for (int kb = 0; kb < 8; ++kb) {
        const _Float16* ap = &sA[(lane & 15) * 256 + kb * 32 + sel * 8];
        if (kb < 7) __builtin_prefetch(brow + (size_t)(kb + 1) * 32 * 16, 0, 3);
        v8h alo = *(const v8h*)(ap);
        v8h ahi = *(const v8h*)(ap + 16);
        v16h a  = __builtin_shufflevector(alo, ahi, 0,1,2,3,4,5,6,7,8,9,10,11,12,13,14,15);
        v16h bm = *(const v16h*)(brow + (size_t)kb * 32 * 16);
        acc = __builtin_amdgcn_wmma_f32_16x16x32_f16(false, a, false, bm,
                                                     (short)0, acc, false, false);
    }
    int col = wave * 16 + (lane & 15);
    #pragma unroll
    for (int r = 0; r < 8; ++r) sout[(sel * 8 + r) * 128 + col] = acc[r];
    __syncthreads();

    // LayerNorm: wave handles rows 2*wave, 2*wave+1
    #pragma unroll
    for (int rr = 0; rr < 2; ++rr) {
        int rowl = wave * 2 + rr;
        float s1 = 0.f, s2 = 0.f;
        #pragma unroll
        for (int cc = lane; cc < 128; cc += 32) {
            float v = sout[rowl * 128 + cc];
            s1 += v; s2 += v * v;
        }
        #pragma unroll
        for (int off = 16; off > 0; off >>= 1) {
            s1 += __shfl_xor(s1, off, 32);
            s2 += __shfl_xor(s2, off, 32);
        }
        float mu  = s1 * (1.0f / 128.0f);
        float var = s2 * (1.0f / 128.0f) - mu * mu;
        float rs  = rsqrtf(var + EPSq);
        int tok   = mtile * 16 + rowl;                 // b*L + l
        int b     = tok >> 14;
        size_t dest = ((size_t)(b << 14) + perm[tok]) * 128;   // final[b, perm[b,l]]
        #pragma unroll
        for (int cc = lane; cc < 128; cc += 32) {
            float v = sout[rowl * 128 + cc];
            dout[dest + cc] = (v - mu) * rs * lnw[cc] + lnb[cc];
        }
    }
}

// =====================================================================
// host launcher
// =====================================================================
extern "C" void kernel_launch(void* const* d_in, const int* in_sizes, int n_in,
                              void* d_out, int out_size, void* d_ws, size_t ws_size,
                              hipStream_t stream) {
    (void)in_sizes; (void)n_in; (void)out_size; (void)ws_size;

    const float* feats  = (const float*)d_in[0];
    const float* pos_w  = (const float*)d_in[1];
    const float* pos_b  = (const float*)d_in[2];
    const float* rms_w  = (const float*)d_in[3];
    const float* inw    = (const float*)d_in[4];   // (128,512)
    const float* convw  = (const float*)d_in[5];   // (256,4)
    const float* convb  = (const float*)d_in[6];
    const float* xprojw = (const float*)d_in[7];   // (256,40)
    const float* dtw    = (const float*)d_in[8];   // (8,256)
    const float* dtb    = (const float*)d_in[9];
    const float* a_log  = (const float*)d_in[10];  // (256,16)
    const float* Dp     = (const float*)d_in[11];
    const float* outw   = (const float*)d_in[12];  // (256,128)
    const float* lnw    = (const float*)d_in[13];
    const float* lnb    = (const float*)d_in[14];
    const int*   coords = (const int*)d_in[15];
    const int*   perm   = (const int*)d_in[16];
    float*       dout   = (float*)d_out;

    char* base = (char*)d_ws;
    size_t off = 0;
    auto carve = [&](size_t bytes) -> char* {
        char* p = base + off;
        off = (off + bytes + 255) & ~(size_t)255;
        return p;
    };
    _Float16* xf16   = (_Float16*)carve((size_t)Nq * Dq * 2);
    _Float16* pBin   = (_Float16*)carve((size_t)32 * 4 * 32 * 16 * 2);   // 512 cols
    _Float16* pBout  = (_Float16*)carve((size_t)8  * 8 * 32 * 16 * 2);   // 128 cols
    float*    aneg   = (float*)   carve((size_t)DIq * DSq * 4);
    float*    xi     = (float*)   carve((size_t)Nq * DIq * 4);
    float*    zbuf   = (float*)   carve((size_t)Nq * DIq * 4);
    float*    xc     = (float*)   carve((size_t)Nq * DIq * 4);
    float*    Bm     = (float*)   carve((size_t)Nq * DSq * 4);
    float*    Cm     = (float*)   carve((size_t)Nq * DSq * 4);
    float*    csa    = (float*)   carve((size_t)Bq * NC * DIq * DSq * 4);
    float*    csh    = (float*)   carve((size_t)Bq * NC * DIq * DSq * 4);
    float*    cstart = (float*)   carve((size_t)Bq * NC * DIq * DSq * 4);
    _Float16* y16    = (_Float16*)carve((size_t)Nq * DIq * 2);
    float*    dt     = xi;   // alias: xi fully consumed by k_conv before k_xproj writes dt

    // weight packing / precompute
    { int tot = 32 * 4 * 32 * 16;
      hipLaunchKernelGGL(k_pack_b, dim3((tot + 255) / 256), dim3(256), 0, stream,
                         inw, pBin, 512, 4, tot); }
    { int tot = 8 * 8 * 32 * 16;
      hipLaunchKernelGGL(k_pack_b, dim3((tot + 255) / 256), dim3(256), 0, stream,
                         outw, pBout, 128, 8, tot); }
    hipLaunchKernelGGL(k_aneg, dim3((DIq * DSq + 255) / 256), dim3(256), 0, stream,
                       a_log, aneg);

    // pipeline
    hipLaunchKernelGGL(k_prep, dim3(Nq), dim3(Dq), 0, stream,
                       feats, pos_w, pos_b, rms_w, coords, perm, xf16);
    hipLaunchKernelGGL(k_gemm_in, dim3(Nq / 16, 4), dim3(256), 0, stream,
                       xf16, pBin, xi, zbuf);
    hipLaunchKernelGGL(k_conv, dim3(Nq), dim3(DIq), 0, stream,
                       xi, convw, convb, xc);
    hipLaunchKernelGGL(k_xproj, dim3(Nq), dim3(DIq), 0, stream,
                       xc, xprojw, dtw, dtb, dt, Bm, Cm);
    hipLaunchKernelGGL(k_scan1, dim3(Bq * NC), dim3(DIq), 0, stream,
                       dt, xc, Bm, aneg, csa, csh);
    hipLaunchKernelGGL(k_scan2, dim3((Bq * DIq * DSq) / 256), dim3(256), 0, stream,
                       csa, csh, cstart);
    hipLaunchKernelGGL(k_scan3, dim3(Bq * NC), dim3(DIq), 0, stream,
                       dt, xc, Bm, Cm, aneg, cstart, Dp, zbuf, y16);
    hipLaunchKernelGGL(k_gemm_out, dim3(Nq / 16), dim3(256), 0, stream,
                       y16, pBout, perm, lnw, lnb, dout);
}
